// GlobalAttentionPool_515396076388
// MI455X (gfx1250) — matrix operand order repaired and checked by reading, not compiled, not run on hardware
//
#include <hip/hip_runtime.h>
#include <hip/hip_bf16.h>

// CDNA5 WMMA vector types
typedef __attribute__((ext_vector_type(16))) __bf16 v16bf;
typedef __attribute__((ext_vector_type(8)))  float  v8f;
typedef __attribute__((ext_vector_type(4)))  float  v4f;

#define N_NODES   100000
#define FDIM      512
#define CDIM      512
#define NSEG      1000
#define ROWS      64                 // rows of X per block (4 M-tiles of 16)
#define KCHUNKS   16                 // FDIM / 32
#define NTILES    32                 // CDIM / 16
#define BLOCK     256                // 8 waves of 32
#define A_ELEMS   (2 * 4 * KCHUNKS * 512)   // parts * mtiles * kchunks * (32 lanes * 16 elems)
#define SMEM_BYTES (A_ELEMS * 2 + ROWS * 4) // 128KB A-frags + 64 segment ids

#define A_PART_STRIDE (4 * KCHUNKS * 512)   // elems between hi and lo A planes
#define A_KC_STRIDE   512                   // elems per kchunk step (A)
#define B_KC_STRIDE   (NTILES * 512)        // elems per kchunk step (B)
#define B_MP_STRIDE   ((size_t)KCHUNKS * NTILES * 512)  // elems per matpart (B)

__global__ void zero_out_kernel(float* __restrict__ out, int n) {
  int i = blockIdx.x * blockDim.x + threadIdx.x;
  if (i < n) out[i] = 0.0f;
}

// fp32 weights -> split bf16 (hi + lo), pre-swizzled into the
// this kernel's contiguous-K B-fragment layout (same K order as the A fragments):
//   lane = N(0..15) + 16*(Kin>=16), e = Kin & 15  (per 16x16 N-tile, 32-K chunk)
// ws layout: [matpart(4: lg_hi, lg_lo, at_hi, at_lo)][kchunk(16)][ntile(32)][lane(32)][e(16)]
__global__ void prep_weights_kernel(const float* __restrict__ lg,
                                    const float* __restrict__ at,
                                    __bf16* __restrict__ ws) {
  int idx = blockIdx.x * blockDim.x + threadIdx.x;   // 0 .. 2*512*512-1
  int mat = idx >> 18;
  int rem = idx & 0x3FFFF;
  int k = rem >> 9;            // 0..511 (F, contraction dim)
  int n = rem & 511;           // 0..511 (C)
  float w = (mat ? at : lg)[(size_t)k * CDIM + n];
  __bf16 hi = (__bf16)w;
  __bf16 lo = (__bf16)(w - (float)hi);
  int kc = k >> 5, kin = k & 31;
  int nt = n >> 4, nin = n & 15;
  int lane = nin + 16 * (kin >> 4);
  int e = kin & 15;
  size_t hiBase = ((size_t)((2 * mat    ) * KCHUNKS + kc) * NTILES + nt) * 512;
  size_t loBase = ((size_t)((2 * mat + 1) * KCHUNKS + kc) * NTILES + nt) * 512;
  ws[hiBase + lane * 16 + e] = hi;
  ws[loBase + lane * 16 + e] = lo;
}

// Epilogue for one 16-row M-tile worth of accumulators held by this lane:
// bias + sigmoid gate, then run-length-merged segment accumulation
// (segment ids are sorted, so one atomic per run instead of per row).
__device__ __forceinline__ void emit_rows(const v8f L, const v8f A,
                                          const int* __restrict__ smSeg,
                                          int mtG, int laneHi,
                                          float bl, float ba,
                                          int& curSeg, float& run,
                                          float* __restrict__ out, int col) {
#pragma unroll
  for (int r = 0; r < 8; ++r) {
    // D layout: VGPR r holds row M=r (lanes 0-15) / M=r+8 (lanes 16-31)
    int s = smSeg[mtG * 16 + laneHi * 8 + r];
    if (s < 0) continue;   // padded rows past N (trailing only)
    float v = (L[r] + bl) * (1.0f / (1.0f + __expf(-(A[r] + ba))));
    if (s != curSeg) {
      if (curSeg >= 0) unsafeAtomicAdd(out + (size_t)curSeg * CDIM + col, run);
      curSeg = s;
      run = v;
    } else {
      run += v;
    }
  }
}

__global__ void __launch_bounds__(BLOCK)
gap_fused_kernel(const float* __restrict__ X, const int* __restrict__ I,
                 const __bf16* __restrict__ W,
                 const float* __restrict__ lg_bias,
                 const float* __restrict__ attn_bias,
                 float* __restrict__ out) {
  extern __shared__ char dynsm[];
  __bf16* smA  = (__bf16*)dynsm;                 // A-fragments, hi/lo split
  int*    smSeg = (int*)(dynsm + A_ELEMS * 2);   // 64 segment ids

  const int tid = threadIdx.x;
  const int r0  = blockIdx.x * ROWS;

  // ---- Phase 1: stage X tile into LDS as swizzled bf16 hi/lo A-fragments ----
  if (tid < ROWS) smSeg[tid] = (r0 + tid < N_NODES) ? I[r0 + tid] : -1;

#pragma clang loop unroll(disable)
  for (int it = 0; it < (ROWS * FDIM) / (BLOCK * 4); ++it) {   // 32 iters
    int li  = (it * BLOCK + tid) * 4;
    int row = li >> 9;           // 0..63
    int k   = li & 511;          // multiple of 4
    int grow = r0 + row;
    v4f x = {0.f, 0.f, 0.f, 0.f};
    if (grow < N_NODES) x = *(const v4f*)(X + (size_t)grow * FDIM + k);
    int mt = row >> 4, m = row & 15;
#pragma unroll
    for (int j = 0; j < 4; ++j) {
      int kk = k + j;
      int kc = kk >> 5, kin = kk & 31;
      // A-fragment swizzle for 16-bit A 16x32:
      //   lanes 0-15 hold K {0..15}, lanes 16-31 hold K {16..31}
      int lane = m + 16 * (kin >> 4);
      int e    = kin & 15;
      float f = x[j];
      __bf16 hi = (__bf16)f;
      __bf16 lo = (__bf16)(f - (float)hi);
      int off = (mt * KCHUNKS + kc) * 512 + lane * 16 + e;
      smA[off] = hi;
      smA[A_PART_STRIDE + off] = lo;
    }
  }
  __syncthreads();

  // ---- Phase 2: WMMA main loops ----
  // Wave w owns M-tiles {2*(w&1), 2*(w&1)+1} and N-tiles (w>>1) + 4*t.
  const int wave   = tid >> 5;
  const int lane   = tid & 31;
  const int laneHi = lane >> 4;
  const int nin    = lane & 15;
  const int mt0    = (wave & 1) * 2;

  // LDS A-fragment base pointers for this wave's two M-tiles (kc = 0)
  const __bf16* aHi0base = smA + (size_t)(mt0    ) * KCHUNKS * 512 + lane * 16;
  const __bf16* aHi1base = smA + (size_t)(mt0 + 1) * KCHUNKS * 512 + lane * 16;

  const v8f vzero = {0.f, 0.f, 0.f, 0.f, 0.f, 0.f, 0.f, 0.f};

#pragma clang loop unroll(disable)
  for (int t = 0; t < 8; ++t) {
    const int nt = (wave >> 1) + 4 * t;       // this iteration's N-tile
    v8f accL0 = vzero, accL1 = vzero, accA0 = vzero, accA1 = vzero;

    const __bf16* wb  = W + (size_t)nt * 512 + lane * 16;   // kc = 0, matpart 0
    const __bf16* ah0 = aHi0base;
    const __bf16* ah1 = aHi1base;

#pragma clang loop unroll(disable)
    for (int kc = 0; kc < KCHUNKS; ++kc) {
      v16bf bLh = *(const v16bf*)(wb + 0 * B_MP_STRIDE);   // lg hi
      v16bf bLl = *(const v16bf*)(wb + 1 * B_MP_STRIDE);   // lg lo
      v16bf bAh = *(const v16bf*)(wb + 2 * B_MP_STRIDE);   // attn hi
      v16bf bAl = *(const v16bf*)(wb + 3 * B_MP_STRIDE);   // attn lo

      v16bf a0h = *(const v16bf*)(ah0);
      v16bf a0l = *(const v16bf*)(ah0 + A_PART_STRIDE);
      // split-bf16: hi*hi + lo*hi + hi*lo (~fp32 accuracy, fp32 accumulate)
      accL0 = __builtin_amdgcn_wmma_f32_16x16x32_bf16(false, a0h, false, bLh, (short)0, accL0, false, false);
      accL0 = __builtin_amdgcn_wmma_f32_16x16x32_bf16(false, a0l, false, bLh, (short)0, accL0, false, false);
      accL0 = __builtin_amdgcn_wmma_f32_16x16x32_bf16(false, a0h, false, bLl, (short)0, accL0, false, false);
      accA0 = __builtin_amdgcn_wmma_f32_16x16x32_bf16(false, a0h, false, bAh, (short)0, accA0, false, false);
      accA0 = __builtin_amdgcn_wmma_f32_16x16x32_bf16(false, a0l, false, bAh, (short)0, accA0, false, false);
      accA0 = __builtin_amdgcn_wmma_f32_16x16x32_bf16(false, a0h, false, bAl, (short)0, accA0, false, false);

      v16bf a1h = *(const v16bf*)(ah1);
      v16bf a1l = *(const v16bf*)(ah1 + A_PART_STRIDE);
      accL1 = __builtin_amdgcn_wmma_f32_16x16x32_bf16(false, a1h, false, bLh, (short)0, accL1, false, false);
      accL1 = __builtin_amdgcn_wmma_f32_16x16x32_bf16(false, a1l, false, bLh, (short)0, accL1, false, false);
      accL1 = __builtin_amdgcn_wmma_f32_16x16x32_bf16(false, a1h, false, bLl, (short)0, accL1, false, false);
      accA1 = __builtin_amdgcn_wmma_f32_16x16x32_bf16(false, a1h, false, bAh, (short)0, accA1, false, false);
      accA1 = __builtin_amdgcn_wmma_f32_16x16x32_bf16(false, a1l, false, bAh, (short)0, accA1, false, false);
      accA1 = __builtin_amdgcn_wmma_f32_16x16x32_bf16(false, a1h, false, bAl, (short)0, accA1, false, false);

      wb  += B_KC_STRIDE;   // next 32-K chunk of weights
      ah0 += A_KC_STRIDE;   // next 32-K chunk of A fragments
      ah1 += A_KC_STRIDE;
    }

    // ---- Epilogue for this N-tile ----
    const int col = nt * 16 + nin;
    const float bl = lg_bias[col];
    const float ba = attn_bias[col];
    int curSeg = -1;
    float run = 0.f;
    emit_rows(accL0, accA0, smSeg, mt0,     laneHi, bl, ba, curSeg, run, out, col);
    emit_rows(accL1, accA1, smSeg, mt0 + 1, laneHi, bl, ba, curSeg, run, out, col);
    if (curSeg >= 0) unsafeAtomicAdd(out + (size_t)curSeg * CDIM + col, run);
  }
}

extern "C" void kernel_launch(void* const* d_in, const int* in_sizes, int n_in,
                              void* d_out, int out_size, void* d_ws, size_t ws_size,
                              hipStream_t stream) {
  const float* X   = (const float*)d_in[0];
  const int*   I   = (const int*)d_in[1];
  const float* lgk = (const float*)d_in[2];
  const float* lgb = (const float*)d_in[3];
  const float* atk = (const float*)d_in[4];
  const float* atb = (const float*)d_in[5];
  float* out = (float*)d_out;
  __bf16* ws = (__bf16*)d_ws;   // 2 MB used: 4 matparts * 512KB

  zero_out_kernel<<<(out_size + BLOCK - 1) / BLOCK, BLOCK, 0, stream>>>(out, out_size);
  prep_weights_kernel<<<(2 * FDIM * CDIM) / BLOCK, BLOCK, 0, stream>>>(lgk, atk, ws);
  int nblocks = (N_NODES + ROWS - 1) / ROWS;   // 1563
  gap_fused_kernel<<<nblocks, BLOCK, SMEM_BYTES, stream>>>(X, I, ws, lgb, atb, out);
}
